// MRConv2d_26044681683387
// MI455X (gfx1250) — matrix operand order repaired and checked by reading, not compiled, not run on hardware
//
#include <hip/hip_runtime.h>
#include <hip/hip_bf16.h>

// ---------------- problem constants (from reference setup) ----------------
#define BATCH 8
#define CH    192      // C
#define CH2   384      // 2C
#define NPIX  3136     // H*W
#define KNN   16
#define NTILE 196      // NPIX/16
#define OTILE 12       // CH/16
#define KBLK  6        // CH/32  (K-blocks of 32 per WMMA pass)

typedef __attribute__((ext_vector_type(16))) __bf16 v16bf;
typedef __attribute__((ext_vector_type(8)))  float  v8f;

union ABFrag { uint4 q[2]; v16bf v; };

// fp32 -> bf16, round-to-nearest-even
__device__ __forceinline__ unsigned short f2bf(float f) {
    unsigned u = __float_as_uint(f);
    u += 0x7FFFu + ((u >> 16) & 1u);
    return (unsigned short)(u >> 16);
}
__device__ __forceinline__ float bflo(unsigned u) {  // low bf16 of dword -> f32
    return __uint_as_float(u << 16);
}
__device__ __forceinline__ float bfhi(unsigned u) {  // high bf16 of dword -> f32
    return __uint_as_float(u & 0xFFFF0000u);
}

// ---------------------------------------------------------------------------
// Kernel 0: pack W (192 x 384 f32, channel-interleaved [x_c, rel_c]) into the
// WMMA *A-matrix* register layout (bf16): even channels -> WeP (multiplies x),
// odd channels -> WoP (multiplies rel).
// A layout (16-bit, ISA 7.12.2): lane L -> row M = L&15 (= output channel),
// half = L>>4; VGPR v holds K pair kl = (v<4 ? half*8+2v : 16+half*8+2(v-4)).
// Entry t = ((kb*OTILE + ot)*32 + lane)*8 + v  => per-lane fragment is 32
// contiguous bytes -> two coalesced b128 loads in the GEMM.
// ---------------------------------------------------------------------------
__global__ void packW_kernel(const float* __restrict__ W,
                             unsigned* __restrict__ WeP,
                             unsigned* __restrict__ WoP) {
    int t = blockIdx.x * blockDim.x + threadIdx.x;   // 18432 total
    if (t >= KBLK * OTILE * 32 * 8) return;
    int v    = t & 7;
    int lane = (t >> 3) & 31;
    int blk  = t >> 8;
    int ot   = blk % OTILE;
    int kb   = blk / OTILE;
    int half = lane >> 4;
    int row  = lane & 15;                 // M row = output channel within tile
    int o    = ot * 16 + row;
    int kl   = (v < 4) ? (half * 8 + 2 * v) : (16 + half * 8 + 2 * (v - 4));
    int k    = kb * 32 + kl;              // channel index (K dim), pair (k,k+1)
    const float* wr = W + (size_t)o * CH2;
    unsigned elo = f2bf(wr[2 * k]);               // W[o, 2k]       (even half)
    unsigned ehi = f2bf(wr[2 * (k + 1)]);
    unsigned olo = f2bf(wr[2 * k + 1]);           // W[o, 2k+1]     (odd half)
    unsigned ohi = f2bf(wr[2 * (k + 1) + 1]);
    WeP[t] = elo | (ehi << 16);
    WoP[t] = olo | (ohi << 16);
}

// ---------------------------------------------------------------------------
// Kernel 1: transpose x (B,C,N) f32 -> xT (B,N,C) bf16 via LDS 32x32 tiles.
// ---------------------------------------------------------------------------
__global__ void transpose_kernel(const float* __restrict__ x,
                                 unsigned short* __restrict__ xT) {
    __shared__ unsigned short tile[32][33];
    int b  = blockIdx.z;
    int n0 = blockIdx.x * 32;          // NPIX/32 = 98 blocks
    int c0 = blockIdx.y * 32;          // CH/32   = 6 blocks
    int tx = threadIdx.x & 31;
    int ty = threadIdx.x >> 5;         // 0..7
    const float* xb = x + (size_t)b * CH * NPIX;
#pragma unroll
    for (int i = 0; i < 32; i += 8) {
        tile[ty + i][tx] = f2bf(xb[(size_t)(c0 + ty + i) * NPIX + n0 + tx]);
    }
    __syncthreads();
    unsigned short* xtb = xT + (size_t)b * NPIX * CH;
#pragma unroll
    for (int i = 0; i < 32; i += 8) {
        xtb[(size_t)(n0 + ty + i) * CH + c0 + tx] = tile[tx][ty + i];
    }
}

// ---------------------------------------------------------------------------
// Kernel 2: max-relative gather. One wave per pixel n; each lane owns 6
// contiguous bf16 channels (12 bytes) so one wave-request covers a full
// 384B gathered feature row (32 lanes x 12B), L2-resident.
//   relT[b,n,c] = max_k ( xT[b, idx0[b,n,k], c] - xT[b, idx1[b,n,k], c] )
// ---------------------------------------------------------------------------
__global__ void maxrel_kernel(const unsigned short* __restrict__ xT,
                              const int* __restrict__ eidx,
                              unsigned short* __restrict__ relT) {
    int gwave = blockIdx.x * 8 + (threadIdx.x >> 5);   // B*N = 25088 waves
    int lane  = threadIdx.x & 31;
    int b = gwave / NPIX;
    int n = gwave - b * NPIX;
    size_t bN = (size_t)b * NPIX;
    const int* e0 = eidx + (bN + (size_t)n) * KNN;                 // x_j
    const int* e1 = e0 + (size_t)BATCH * NPIX * KNN;               // x_i
    // gathered-row byte offsets (buffer < 2^32 bytes, keep 32-bit)
    unsigned offj[KNN], offi[KNN];
#pragma unroll
    for (int k = 0; k < KNN; ++k) {
        offj[k] = ((unsigned)bN + (unsigned)e0[k]) * (CH * 2);
        offi[k] = ((unsigned)bN + (unsigned)e1[k]) * (CH * 2);
    }
    const char* basep = (const char*)xT;
    int co = lane * 12;                 // lane's 6 channels (3 dwords)
    float m0 = -3.4e38f, m1 = m0, m2 = m0, m3 = m0, m4 = m0, m5 = m0;
#pragma unroll
    for (int k = 0; k < KNN; ++k) {
        const unsigned* pj = (const unsigned*)(basep + (offj[k] + co));
        const unsigned* pi = (const unsigned*)(basep + (offi[k] + co));
        unsigned j0 = pj[0], j1 = pj[1], j2 = pj[2];   // merged -> b96
        unsigned i0 = pi[0], i1 = pi[1], i2 = pi[2];
        m0 = fmaxf(m0, bflo(j0) - bflo(i0));
        m1 = fmaxf(m1, bfhi(j0) - bfhi(i0));
        m2 = fmaxf(m2, bflo(j1) - bflo(i1));
        m3 = fmaxf(m3, bfhi(j1) - bfhi(i1));
        m4 = fmaxf(m4, bflo(j2) - bflo(i2));
        m5 = fmaxf(m5, bfhi(j2) - bfhi(i2));
    }
    unsigned* q = (unsigned*)((char*)relT + (bN + (size_t)n) * (CH * 2) + co);
    q[0] = f2bf(m0) | ((unsigned)f2bf(m1) << 16);
    q[1] = f2bf(m2) | ((unsigned)f2bf(m3) << 16);
    q[2] = f2bf(m4) | ((unsigned)f2bf(m5) << 16);
}

// ---------------------------------------------------------------------------
// Kernel 3: WMMA GEMM with A = weights, B = features, so D rows = output
// channels and D columns = pixels => fully coalesced 64B output stores and
// bias loads uniform per 16-lane half.
//   out[b,o,n] = relu( sum_c We[o,c]*xT[b,n,c] + Wo[o,c]*relT[b,n,c] + bias[o] )
// B fragment (16-bit dense layout): lane col = pixel = L&15, elements
// e=0..7 are channels kb*32 + half*8 + e, e=8..15 are kb*32 + 16 + half*8 + (e-8)
// -> two b128 loads from the row-major (N,C) bf16 buffers.
// ---------------------------------------------------------------------------
__global__ void gemm_kernel(const unsigned short* __restrict__ xT,
                            const unsigned short* __restrict__ relT,
                            const unsigned* __restrict__ WeP,
                            const unsigned* __restrict__ WoP,
                            const float* __restrict__ bias,
                            float* __restrict__ out) {
    int lane = threadIdx.x & 31;
    int tile = blockIdx.x * 8 + (threadIdx.x >> 5);    // 18816 tiles, exact fit
    int ot = tile % OTILE;
    int t2 = tile / OTILE;
    int nt = t2 % NTILE;
    int b  = t2 / NTILE;

    int n0 = nt * 16, o0 = ot * 16;
    int half = lane >> 4;
    int col  = lane & 15;               // pixel column of the tile
    size_t bN = (size_t)b * NPIX;

    const unsigned short* brow_x = xT   + (bN + n0 + col) * CH + half * 8;
    const unsigned short* brow_r = relT + (bN + n0 + col) * CH + half * 8;

    v8f acc = {};
#pragma unroll
    for (int kb = 0; kb < KBLK; ++kb) {
        ABFrag ae, ao, bx, br;
        const uint4* pe = (const uint4*)(WeP + (((size_t)kb * OTILE + ot) * 32 + lane) * 8);
        ae.q[0] = pe[0]; ae.q[1] = pe[1];
        const uint4* po = (const uint4*)(WoP + (((size_t)kb * OTILE + ot) * 32 + lane) * 8);
        ao.q[0] = po[0]; ao.q[1] = po[1];
        const unsigned short* px = brow_x + kb * 32;
        bx.q[0] = *(const uint4*)(px);
        bx.q[1] = *(const uint4*)(px + 16);
        const unsigned short* pr = brow_r + kb * 32;
        br.q[0] = *(const uint4*)(pr);
        br.q[1] = *(const uint4*)(pr + 16);

        acc = __builtin_amdgcn_wmma_f32_16x16x32_bf16(
                  false, ae.v, false, bx.v, (short)0, acc, false, false);
        acc = __builtin_amdgcn_wmma_f32_16x16x32_bf16(
                  false, ao.v, false, br.v, (short)0, acc, false, false);
    }

    // D layout: col = lane&15 (pixel), VGPR r -> M = r + 8*half (out channel)
    size_t orow = ((size_t)b * CH + o0 + 8 * half) * NPIX + n0 + col;
#pragma unroll
    for (int r = 0; r < 8; ++r) {
        float bv = bias[o0 + 8 * half + r];            // uniform within each 16-lane half
        out[orow + (size_t)r * NPIX] = fmaxf(acc[r] + bv, 0.0f);
    }
}

// ---------------------------------------------------------------------------
// Host-side launch. Inputs: d_in[0]=x f32 (B,C,H,W), d_in[1]=edge_index i32
// (2,B,N,K), d_in[2]=W f32 (C,2C), d_in[3]=b f32 (C). Output f32 (B,C,H,W).
// Workspace: xT bf16 (9.19MB) + relT bf16 (9.19MB) + 2x packed W (72KB).
// ---------------------------------------------------------------------------
extern "C" void kernel_launch(void* const* d_in, const int* in_sizes, int n_in,
                              void* d_out, int out_size, void* d_ws, size_t ws_size,
                              hipStream_t stream) {
    const float* x    = (const float*)d_in[0];
    const int*   eidx = (const int*)d_in[1];
    const float* W    = (const float*)d_in[2];
    const float* bias = (const float*)d_in[3];
    float*       out  = (float*)d_out;

    const size_t XT_BYTES = (size_t)BATCH * NPIX * CH * sizeof(unsigned short); // 9,633,792
    const size_t WP_BYTES = (size_t)KBLK * OTILE * 32 * 8 * sizeof(unsigned);   // 73,728
    char* ws = (char*)d_ws;
    unsigned short* xT   = (unsigned short*)(ws);
    unsigned short* relT = (unsigned short*)(ws + XT_BYTES);
    unsigned*       WeP  = (unsigned*)(ws + 2 * XT_BYTES);
    unsigned*       WoP  = (unsigned*)(ws + 2 * XT_BYTES + WP_BYTES);

    // 0) pack weights into WMMA A-fragment layout (bf16, even/odd channels)
    packW_kernel<<<72, 256, 0, stream>>>(W, WeP, WoP);

    // 1) transpose + bf16 convert: (B,C,N) -> (B,N,C)
    transpose_kernel<<<dim3(NPIX / 32, CH / 32, BATCH), 256, 0, stream>>>(x, xT);

    // 2) gathered max-relative features: one wave per pixel, full-row requests
    maxrel_kernel<<<(BATCH * NPIX) / 8, 256, 0, stream>>>(xT, eidx, relT);

    // 3) fused dual-pass WMMA GEMM + bias + ReLU, coalesced stores
    gemm_kernel<<<(BATCH * NTILE * OTILE) / 8, 256, 0, stream>>>(
        xT, relT, WeP, WoP, bias, out);
}
